// RNN_71717363908810
// MI455X (gfx1250) — matrix operand order repaired and checked by reading, not compiled, not run on hardware
//
#include <hip/hip_runtime.h>
#include <hip/hip_bf16.h>

// ---------------------------------------------------------------------------
// Elman RNN on gfx1250 (wave32, WMMA).
//   pre[b,t,:]  = emb[x[b,t]] @ Wx + b1        (parallel, WMMA bf16)
//   h_t         = sigmoid(pre_t + h_{t-1}@Wh)  (8 persistent WGs, Wh in LDS)
//   out[b,t,:]  = h_t @ (W2@W3) + (b2@W3+b3)   (parallel, VALU + shfl reduce)
// ---------------------------------------------------------------------------

typedef __attribute__((ext_vector_type(16))) __bf16 v16bf;
typedef __attribute__((ext_vector_type(8)))  float  v8f;

#define VOCAB 32000
#define EMB   300
#define HID   300
#define HP    320            // padded hidden/emb (20 tiles of 16, 10 chunks of 32)
#define NTL   20             // N tiles = HP/16
#define KCN   10             // K chunks = HP/32
#define BB    128
#define TT    512
#define NTOK  (BB*TT)
#define MT    (NTOK/16)      // 4096 token tiles; tile = t*8 + b0/16

// workspace layout (bytes). pre tile is consumed at step t and overwritten
// in-place by h at step t, so one buffer serves both.
#define SZ_PREH   ((size_t)MT*NTL*32*8*2)     // 41,943,040 (bf16, C-layout tiles)
#define SZ_WSWZ   ((size_t)KCN*NTL*32*16*2)   // 204,800 per matrix
#define OFF_PREH  0ull
#define OFF_WX    (OFF_PREH + SZ_PREH)
#define OFF_WH    (OFF_WX + SZ_WSWZ)
#define OFF_W23   (OFF_WH + SZ_WSWZ)
#define OFF_B23   (OFF_W23 + (size_t)HP*2*4)

__device__ __forceinline__ unsigned short f2bf(float f) {
    union { float f; unsigned u; } v; v.f = f;
    unsigned r = v.u + 0x7FFFu + ((v.u >> 16) & 1u);   // RNE
    return (unsigned short)(r >> 16);
}
__device__ __forceinline__ float bf2f(unsigned h) {
    union { unsigned u; float f; } v; v.u = (h & 0xFFFFu) << 16;
    return v.f;
}
__device__ __forceinline__ uint4 pack8(const float* f) {
    uint4 o;
    o.x = (unsigned)f2bf(f[0]) | ((unsigned)f2bf(f[1]) << 16);
    o.y = (unsigned)f2bf(f[2]) | ((unsigned)f2bf(f[3]) << 16);
    o.z = (unsigned)f2bf(f[4]) | ((unsigned)f2bf(f[5]) << 16);
    o.w = (unsigned)f2bf(f[6]) | ((unsigned)f2bf(f[7]) << 16);
    return o;
}
__device__ __forceinline__ void unpack8(uint4 v, float* f) {
    f[0]=bf2f(v.x); f[1]=bf2f(v.x>>16); f[2]=bf2f(v.y); f[3]=bf2f(v.y>>16);
    f[4]=bf2f(v.z); f[5]=bf2f(v.z>>16); f[6]=bf2f(v.w); f[7]=bf2f(v.w>>16);
}

// --- swizzle helpers matching the 16-bit WMMA VGPR layouts (per-lane 32B) ---
// B (KxN=32x16): lane n + 16*h, idx e: K = chunk + 8*h + (e&7) + 16*(e>>3)
// A (MxN=16x32): element (m, rel-k r): lane = m + 16*((r>>3)&1), idx = (r&7)+8*(r>>4)
__device__ __forceinline__ int a_swz(int m, int k) {      // -> half index in [KCN*32*16]
    int kc = k >> 5, r = k & 31;
    int lane = m + (((r >> 3) & 1) << 4);
    int e = (r & 7) + ((r >> 4) << 3);
    return (kc * 32 + lane) * 16 + e;
}

// ---------------------------------------------------------------------------
// Kernel 0a: build bf16 swizzled Wx / Wh (contiguous) from W1[600][300].
// ---------------------------------------------------------------------------
__global__ __launch_bounds__(256) void prep_w1(const float* __restrict__ W1,
                                               unsigned short* __restrict__ dst) {
    const int total = KCN * NTL * 32 * 16;                // per matrix
    int idx = blockIdx.x * blockDim.x + threadIdx.x;
    if (idx >= 2 * total) return;
    int which = idx / total;                              // 0=Wx, 1=Wh
    int r   = idx % total;
    int e   = r & 15;
    int lane = (r >> 4) & 31;
    int rec = r >> 9;
    int nt  = rec % NTL;
    int kc  = rec / NTL;
    int k   = kc * 32 + ((lane >> 4) << 3) + (e & 7) + ((e >> 3) << 4);
    int n   = nt * 16 + (lane & 15);
    float v = 0.f;
    if (k < 300 && n < HID) v = W1[(which * 300 + k) * HID + n];
    dst[idx] = f2bf(v);
}

// ---------------------------------------------------------------------------
// Kernel 0b: fold output head: W23[HP][2] = W2@W3 (zero-padded), b23 = b2@W3+b3
// ---------------------------------------------------------------------------
__global__ __launch_bounds__(HP) void prep_w23(const float* __restrict__ W2,
                                               const float* __restrict__ b2,
                                               const float* __restrict__ W3,
                                               const float* __restrict__ b3,
                                               float* __restrict__ W23,
                                               float* __restrict__ b23) {
    int n = threadIdx.x;
    float s0 = 0.f, s1 = 0.f;
    if (n < HID)
        for (int j = 0; j < HID; ++j) {
            float w = W2[n * HID + j];
            s0 += w * W3[j * 2 + 0];
            s1 += w * W3[j * 2 + 1];
        }
    W23[n * 2 + 0] = s0;
    W23[n * 2 + 1] = s1;
    if (n == 0) {
        float t0 = b3[0], t1 = b3[1];
        for (int j = 0; j < HID; ++j) {
            t0 += b2[j] * W3[j * 2 + 0];
            t1 += b2[j] * W3[j * 2 + 1];
        }
        b23[0] = t0; b23[1] = t1;
    }
}

// ---------------------------------------------------------------------------
// Kernel 1: x-projection. One WG (320 thr = 10 waves) per 16-token tile.
// pre[tile][nt] stored in WMMA C-layout: 32 lanes x 8 bf16 (one b128/lane).
// ---------------------------------------------------------------------------
__global__ __launch_bounds__(320) void xproj(const int* __restrict__ x,
                                             const float* __restrict__ emb,
                                             const float* __restrict__ b1,
                                             const unsigned short* __restrict__ wxs,
                                             unsigned short* __restrict__ preS) {
    __shared__ __align__(16) unsigned short aswz[KCN * 32 * 16];   // 10 KB
    const int mtile = blockIdx.x;
    const int tid = threadIdx.x, lane = tid & 31, wid = tid >> 5;

    // stage 16 embedding rows (token-gather, fp32->bf16) into A-swizzle
    for (int m = 0; m < 16; ++m) {
        int token = mtile * 16 + m;                 // token = t*B + b
        int b = token & (BB - 1), t = token >> 7;
        int k = tid;                                // 0..319
        float v = 0.f;
        if (k < EMB) {
            int row = x[b * TT + t];
            v = emb[(size_t)row * EMB + k];
        }
        aswz[a_swz(m, k)] = f2bf(v);
    }
    __syncthreads();

    for (int w = 0; w < 2; ++w) {
        int nt = wid + w * 10;
        v8f acc = {};
        for (int kc = 0; kc < KCN; ++kc) {
            v16bf a  = *(const v16bf*)&aswz[(kc * 32 + lane) * 16];
            v16bf bm = *(const v16bf*)&wxs[(size_t)((kc * NTL + nt) * 32 + lane) * 16];
            acc = __builtin_amdgcn_wmma_f32_16x16x32_bf16(
                false, a, false, bm, (short)0, acc, false, false);
        }
        int n = nt * 16 + (lane & 15);
        float bias = (n < HID) ? b1[n] : 0.f;
        float f[8];
        for (int v = 0; v < 8; ++v) f[v] = acc[v] + bias;
        ((uint4*)preS)[((size_t)mtile * NTL + nt) * 32 + lane] = pack8(f);
    }
}

// ---------------------------------------------------------------------------
// Kernel 2: recurrence. 8 persistent WGs (one per 16-row batch tile).
// Wh (200 KB bf16) + double-buffered h live in LDS (dynamic, 225 KB < 320 KB).
// Per step: 10 waves x 2 N-tiles x 10 K-chunks of v_wmma_f32_16x16x32_bf16.
// h overwrites pre in-place (same tile, same C-layout) for kernel 3.
// ---------------------------------------------------------------------------
__global__ __launch_bounds__(320) void recurrence(unsigned short* __restrict__ preh,
                                                  const unsigned short* __restrict__ whs) {
    extern __shared__ __align__(16) char smem[];
    unsigned short* whl  = (unsigned short*)smem;                      // 204800 B
    unsigned short* hbuf = (unsigned short*)(smem + SZ_WSWZ);          // 2 x 10240 B
    const int tid = threadIdx.x, lane = tid & 31, wid = tid >> 5;
    const int wg = blockIdx.x;                                         // 0..7

    // stage Wh into LDS; zero h buffer 0 (h_{-1} = 0)
    {
        const uint4* src = (const uint4*)whs;
        uint4* dst = (uint4*)whl;
        for (int i = tid; i < (KCN * NTL * 32 * 16) / 8; i += 320) dst[i] = src[i];
        uint4 z = {0u, 0u, 0u, 0u};
        uint4* hz = (uint4*)hbuf;
        for (int i = tid; i < (KCN * 32 * 16) / 8; i += 320) hz[i] = z;
    }
    __syncthreads();

    int cur = 0;
    for (int t = 0; t < TT; ++t) {
        const int tile = t * 8 + wg;
        unsigned short* hc = hbuf + cur * (KCN * 32 * 16);
        unsigned short* hn = hbuf + (cur ^ 1) * (KCN * 32 * 16);

        for (int w = 0; w < 2; ++w) {
            int nt = wid + w * 10;
            size_t prec = ((size_t)tile * NTL + nt) * 32 + lane;

            // prefetch next timestep's pre tile while this chain runs
            if (t + 1 < TT)
                __builtin_prefetch(&((const uint4*)preh)[prec + (size_t)8 * NTL * 32], 0, 0);

            uint4 pv = ((const uint4*)preh)[prec];
            float f[8];
            unpack8(pv, f);
            v8f acc;
            for (int v = 0; v < 8; ++v) acc[v] = f[v];

            for (int kc = 0; kc < KCN; ++kc) {
                v16bf a  = *(const v16bf*)&hc[(kc * 32 + lane) * 16];
                v16bf bm = *(const v16bf*)&whl[((kc * NTL + nt) * 32 + lane) * 16];
                acc = __builtin_amdgcn_wmma_f32_16x16x32_bf16(
                    false, a, false, bm, (short)0, acc, false, false);
            }

            // sigmoid; store h (overwrites pre tile) + scatter into A-swizzle
            unsigned short hb[8];
            float hf[8];
            for (int v = 0; v < 8; ++v) {
                float s = 1.0f / (1.0f + __expf(-acc[v]));
                hf[v] = s;
                hb[v] = f2bf(s);
            }
            ((uint4*)preh)[prec] = pack8(hf);

            int n = nt * 16 + (lane & 15);
            int kc2 = n >> 5, r = n & 31;
            int ei = (r & 7) + ((r >> 4) << 3);
            int lphi = ((r >> 3) & 1) << 4;
            int mbase = (lane >> 4) << 3;
            for (int v = 0; v < 8; ++v)
                hn[(kc2 * 32 + (mbase + v + lphi)) * 16 + ei] = hb[v];
        }
        __syncthreads();        // writes to hn visible before next step reads
        cur ^= 1;
    }
}

// ---------------------------------------------------------------------------
// Kernel 3: output head. One wave per tile: coalesced b128 h loads, per-lane
// partial dots against W23, 16-lane shfl_xor reduction, float2 stores.
// ---------------------------------------------------------------------------
__global__ __launch_bounds__(256) void outproj(const unsigned short* __restrict__ hS,
                                               const float* __restrict__ W23,
                                               const float* __restrict__ b23,
                                               float* __restrict__ out) {
    const int tid = threadIdx.x, lane = tid & 31, wid = tid >> 5;
    const int gtile = blockIdx.x * 8 + wid;                  // 0..4095
    const int nsub = lane & 15;
    float p0[8], p1[8];
    for (int v = 0; v < 8; ++v) { p0[v] = 0.f; p1[v] = 0.f; }

    for (int nt = 0; nt < NTL; ++nt) {
        uint4 hv = ((const uint4*)hS)[((size_t)gtile * NTL + nt) * 32 + lane];
        int n = nt * 16 + nsub;                              // padded rows of W23 are 0
        float w0 = W23[n * 2 + 0], w1 = W23[n * 2 + 1];
        float h[8];
        unpack8(hv, h);
        for (int v = 0; v < 8; ++v) { p0[v] += h[v] * w0; p1[v] += h[v] * w1; }
    }
    for (int v = 0; v < 8; ++v)
        for (int s = 1; s < 16; s <<= 1) {
            p0[v] += __shfl_xor(p0[v], s, 32);
            p1[v] += __shfl_xor(p1[v], s, 32);
        }
    if (nsub == 0) {
        int t = gtile >> 3, b0 = (gtile & 7) * 16, mb = (lane >> 4) * 8;
        float bb0 = b23[0], bb1 = b23[1];
        for (int v = 0; v < 8; ++v) {
            int b = b0 + mb + v;
            float2 o; o.x = p0[v] + bb0; o.y = p1[v] + bb1;
            ((float2*)out)[(size_t)b * TT + t] = o;
        }
    }
}

// ---------------------------------------------------------------------------
extern "C" void kernel_launch(void* const* d_in, const int* in_sizes, int n_in,
                              void* d_out, int out_size, void* d_ws, size_t ws_size,
                              hipStream_t stream) {
    const int*   x   = (const int*)d_in[0];
    const float* emb = (const float*)d_in[1];
    const float* W1  = (const float*)d_in[2];
    const float* b1  = (const float*)d_in[3];
    const float* W2  = (const float*)d_in[4];
    const float* b2  = (const float*)d_in[5];
    const float* W3  = (const float*)d_in[6];
    const float* b3  = (const float*)d_in[7];
    char* ws = (char*)d_ws;
    unsigned short* preh = (unsigned short*)(ws + OFF_PREH);
    unsigned short* wswz = (unsigned short*)(ws + OFF_WX);   // Wx then Wh contiguous
    unsigned short* whs  = (unsigned short*)(ws + OFF_WH);
    float* W23  = (float*)(ws + OFF_W23);
    float* b23v = (float*)(ws + OFF_B23);
    float* out  = (float*)d_out;

    int prepN = 2 * KCN * NTL * 32 * 16;
    prep_w1 <<<(prepN + 255) / 256, 256, 0, stream>>>(W1, wswz);
    prep_w23<<<1, HP, 0, stream>>>(W2, b2, W3, b3, W23, b23v);
    xproj   <<<MT, 320, 0, stream>>>(x, emb, b1, wswz, preh);
    size_t ldsB = SZ_WSWZ + 2 * (size_t)KCN * 32 * 16 * 2;   // 225,280 B dynamic LDS
    recurrence<<<8, 320, ldsB, stream>>>(preh, whs);
    outproj <<<MT / 8, 256, 0, stream>>>(preh, W23, b23v, out);
}
